// SelfAttention_72782515798598
// MI455X (gfx1250) — matrix operand order repaired and checked by reading, not compiled, not run on hardware
//
#include <hip/hip_runtime.h>
#include <hip/hip_bf16.h>
#include <math.h>

// ---------------------------------------------------------------------------
// Transformer encoder layer for MI455X (gfx1250), wave32 + WMMA bf16.
//   B=4, S=2048, E=1024, H=16, HD=64, FFN=4096
// ---------------------------------------------------------------------------

typedef __attribute__((ext_vector_type(8)))  __bf16 v8bf;
typedef __attribute__((ext_vector_type(16))) __bf16 v16bf;
typedef __attribute__((ext_vector_type(8)))  float  v8f;
typedef __attribute__((ext_vector_type(4)))  int    v4i_t;

#define EMBED 1024
#define HEADS 16
#define HDIM  64
#define FFND  4096
#define BB    4
#define SS    2048
#define MROWS (BB * SS)   // 8192 token rows

static __device__ __forceinline__ unsigned short f2bf(float f) {
  unsigned u = __float_as_uint(f);
  u += 0x7FFFu + ((u >> 16) & 1u);          // round-to-nearest-even
  return (unsigned short)(u >> 16);
}
static __device__ __forceinline__ v8bf ld8(const unsigned short* p) {
  union { uint4 u; v8bf b; } t;
  t.u = *(const uint4*)p;                    // 16B load
  return t.b;
}
static __device__ __forceinline__ v16bf mk16(v8bf lo, v8bf hi) {
  return __builtin_shufflevector(lo, hi, 0,1,2,3,4,5,6,7,8,9,10,11,12,13,14,15);
}
static __device__ __forceinline__ v8f zero8() {
  v8f z;
#pragma unroll
  for (int i = 0; i < 8; ++i) z[i] = 0.f;
  return z;
}

// ---- async global -> LDS (ASYNCcnt path), guarded for toolchain support ----
// Prototype (from clang diagnostic): (int4 __device__*, int4 __shared__*, Ii, Ii)
#if __has_builtin(__builtin_amdgcn_global_load_async_to_lds_b128)
#define ASYNC_LDS 1
typedef __attribute__((address_space(1))) v4i_t gv4i_t;   // global int4
typedef __attribute__((address_space(3))) v4i_t lv4i_t;   // LDS int4
static __device__ __forceinline__ void async_b128(const unsigned short* g,
                                                  unsigned short* l) {
  __builtin_amdgcn_global_load_async_to_lds_b128((gv4i_t*)g, (lv4i_t*)l, 0, 0);
}
static __device__ __forceinline__ void wait_async() {
#if __has_builtin(__builtin_amdgcn_s_wait_asynccnt)
  __builtin_amdgcn_s_wait_asynccnt(0);
#else
  asm volatile("s_wait_asynccnt 0x0" ::: "memory");
#endif
}
#else
#define ASYNC_LDS 0
static __device__ __forceinline__ void wait_async() {}
#endif

// Width-16 XOR shuffle on the VALU via v_permlane16_b32 (keeps the DS pipe
// free for WMMA fragment gathers). Falls back to __shfl_xor (ds_bpermute).
template<int M>
static __device__ __forceinline__ float xor16(float x) {
#if __has_builtin(__builtin_amdgcn_permlane16)
  constexpr int s0 = (M == 1) ? 0x67452301
                   : (M == 2) ? 0x54761032
                   : (M == 4) ? 0x32107654
                   :            (int)0xFEDCBA98;
  constexpr int s1 = (M == 1) ? (int)0xEFCDAB89
                   : (M == 2) ? (int)0xDCFE98BA
                   : (M == 4) ? (int)0xBA98FEDC
                   :            0x76543210;
  int xi = __float_as_int(x);
  return __int_as_float(__builtin_amdgcn_permlane16(xi, xi, s0, s1, false, false));
#else
  return __shfl_xor(x, M, 16);
#endif
}

// --------------------------- fp32 -> bf16 copy -----------------------------
__global__ __launch_bounds__(256) void cvt_bf16(const float* __restrict__ in,
                                                unsigned short* __restrict__ out,
                                                int n) {
  int i = blockIdx.x * 256 + threadIdx.x;
  if (i < n) out[i] = f2bf(in[i]);
}

// --------------- fp32 [K][N] -> bf16 [N][K] (weight transpose) -------------
__global__ __launch_bounds__(256) void transpose_to_bf16(
    const float* __restrict__ Win, unsigned short* __restrict__ Wt,
    int K, int N) {
  __shared__ float tile[32][33];
  const int t = threadIdx.x, tx = t & 31, ty = t >> 5;
  const int k0 = blockIdx.y * 32, n0 = blockIdx.x * 32;
#pragma unroll
  for (int i = 0; i < 4; ++i)
    tile[ty + i * 8][tx] = Win[(size_t)(k0 + ty + i * 8) * N + n0 + tx];
  __syncthreads();
#pragma unroll
  for (int i = 0; i < 4; ++i)
    Wt[(size_t)(n0 + ty + i * 8) * K + k0 + tx] = f2bf(tile[tx][ty + i * 8]);
}

// ------------------------------ WMMA GEMM ----------------------------------
// C[M][N] = act( A[M][K](bf16) * Wt[N][K]^T(bf16) + bias )   (f32 accumulate)
// Block tile 128x256, 8 waves (2x4), wave tile 64x64 -> 16 wmma / K-step.
// Double-buffered LDS; tile k+1 staged via global_load_async_to_lds_b128
// (ASYNCcnt) while tile k runs its 16 WMMAs.
// ACT: 0 = none, 1 = exact GELU.  OBF: 1 = bf16 out, 0 = f32 out.
template<int ACT, int OBF>
__global__ __launch_bounds__(256) void gemm_wmma(
    const unsigned short* __restrict__ A,
    const unsigned short* __restrict__ Wt,
    const float* __restrict__ bias,
    float* __restrict__ outF,
    unsigned short* __restrict__ outB,
    int M, int N, int K) {
  constexpr int LDT = 40;                    // 32 + 8 pad (multiple of 8)
  __shared__ __align__(16) unsigned short Asm[2][128 * LDT];
  __shared__ __align__(16) unsigned short Bsm[2][256 * LDT];

  const int t = threadIdx.x;
  const int w = t >> 5, lane = t & 31, lrow = lane & 15;
  const bool hi = lane >= 16;
  const int wm = w >> 2, wn = w & 3;         // 2 x 4 wave grid
  const int m0 = blockIdx.y * 128, n0 = blockIdx.x * 256;
  const int ar = t >> 1, acol = (t & 1) * 16; // A copy coords (2x b128)

  v8f acc[4][4];
#pragma unroll
  for (int im = 0; im < 4; ++im)
#pragma unroll
    for (int in = 0; in < 4; ++in) acc[im][in] = zero8();

  auto stage = [&](int buf, int k0) {
    const unsigned short* ga = A  + (size_t)(m0 + ar) * K + k0 + acol;
    const unsigned short* gb = Wt + (size_t)(n0 + t) * K + k0;
    unsigned short* la = &Asm[buf][ar * LDT + acol];
    unsigned short* lb = &Bsm[buf][t * LDT];
#if ASYNC_LDS
    async_b128(ga,      la);
    async_b128(ga + 8,  la + 8);
    async_b128(gb,      lb);
    async_b128(gb + 8,  lb + 8);
    async_b128(gb + 16, lb + 16);
    async_b128(gb + 24, lb + 24);
#else
    const uint4* g4a = (const uint4*)ga;
    uint4 a0 = g4a[0], a1 = g4a[1];
    const uint4* g4b = (const uint4*)gb;
    uint4 b0 = g4b[0], b1 = g4b[1], b2 = g4b[2], b3 = g4b[3];
    *(uint4*)(la)      = a0;  *(uint4*)(la + 8)  = a1;
    *(uint4*)(lb)      = b0;  *(uint4*)(lb + 8)  = b1;
    *(uint4*)(lb + 16) = b2;  *(uint4*)(lb + 24) = b3;
#endif
  };

  stage(0, 0);
  const int nst = K >> 5;
  for (int si = 0; si < nst; ++si) {
    const int cur = si & 1;
    wait_async();           // this wave's async tile writes have landed
    __syncthreads();        // everyone's tile is in LDS
    if (si + 1 < nst) stage(cur ^ 1, (si + 1) << 5);

    // A fragment (16-bit 16x32): lane<16 row=lrow holds K 0..15;
    // lane>=16 holds K 16..31.
    const int c0 = hi ? 16 : 0;
    // B fragment (32x16): lane<16 col=lrow holds K 0..15; lane>=16 K 16..31.
    const int bc = hi ? 16 : 0;
    v16bf af[4], bfr[4];
#pragma unroll
    for (int im = 0; im < 4; ++im) {
      int row = wm * 64 + im * 16 + lrow;
      af[im] = mk16(ld8(&Asm[cur][row * LDT + c0]),
                    ld8(&Asm[cur][row * LDT + c0 + 8]));
    }
#pragma unroll
    for (int in = 0; in < 4; ++in) {
      int row = wn * 64 + in * 16 + lrow;
      bfr[in] = mk16(ld8(&Bsm[cur][row * LDT + bc]),
                     ld8(&Bsm[cur][row * LDT + bc + 8]));
    }
#pragma unroll
    for (int im = 0; im < 4; ++im)
#pragma unroll
      for (int in = 0; in < 4; ++in)
        acc[im][in] = __builtin_amdgcn_wmma_f32_16x16x32_bf16(
            false, af[im], false, bfr[in], (short)0, acc[im][in], false, false);
    __syncthreads();
  }

  // Epilogue: C layout puts row r (lane<16) / r+8 (lane>=16), col = lane&15.
#pragma unroll
  for (int im = 0; im < 4; ++im)
#pragma unroll
    for (int in = 0; in < 4; ++in) {
      int col = n0 + wn * 64 + in * 16 + lrow;
      float bsv = bias[col];
#pragma unroll
      for (int r = 0; r < 8; ++r) {
        int row = m0 + wm * 64 + im * 16 + r + (hi ? 8 : 0);
        float vv = acc[im][in][r] + bsv;
        if (ACT == 1) vv = 0.5f * vv * (1.f + erff(vv * 0.70710678118654752f));
        if (OBF) outB[(size_t)row * N + col] = f2bf(vv);
        else     outF[(size_t)row * N + col] = vv;
      }
    }
}

// --------------------------- Flash attention -------------------------------
// grid = (S/128, H, B), block = 256 (8 waves). Wave owns a 16-row q stripe
// with its q fragments resident in VGPRs. kv tiled by 64: 8 score wmma +
// 8 P*V wmma per staging round; softmax stats amortized over 64 kv columns.
__global__ __launch_bounds__(256) void flash_attn(
    const unsigned short* __restrict__ q,
    const unsigned short* __restrict__ k,
    const unsigned short* __restrict__ v,
    unsigned short* __restrict__ ctx) {
  __shared__ __align__(16) unsigned short kT[64 * 72];     // [kv][64+8]
  __shared__ __align__(16) unsigned short vT[64 * 72];     // [d][64+8] (transposed)
  __shared__ __align__(16) unsigned short pS[8 * 16 * 72]; // per-wave P tiles

  const int t = threadIdx.x, w = t >> 5, lane = t & 31, lrow = lane & 15;
  const bool hi = lane >= 16;
  const int b = blockIdx.z, h = blockIdx.y;
  const int q0 = blockIdx.x * 128 + w * 16;
  const size_t hoff = (size_t)h * HDIM;
  const int c0 = hi ? 16 : 0, bc = hi ? 16 : 0;

  // q fragments stay in VGPRs for the whole kv sweep (A-matrix layout).
  const unsigned short* qrp = q + ((size_t)b * SS + q0 + lrow) * EMBED + hoff;
  v16bf aq[2];
  aq[0] = mk16(ld8(qrp + c0),      ld8(qrp + c0 + 8));
  aq[1] = mk16(ld8(qrp + 32 + c0), ld8(qrp + 32 + c0 + 8));

  v8f o[4];
#pragma unroll
  for (int dt = 0; dt < 4; ++dt) o[dt] = zero8();
  float mr[8], lr[8];
#pragma unroll
  for (int r = 0; r < 8; ++r) { mr[r] = -3.0e38f; lr[r] = 0.f; }

  for (int j = 0; j < SS; j += 64) {
    __syncthreads();
    // cooperative stage: k tile row-major (async path), v transposed via VGPR.
    {
      int r = t >> 2, seg = (t & 3) * 16;
      const unsigned short* kp = k + ((size_t)b * SS + j + r) * EMBED + hoff + seg;
#if ASYNC_LDS
      async_b128(kp,     &kT[r * 72 + seg]);
      async_b128(kp + 8, &kT[r * 72 + seg + 8]);
#else
      uint4 k0v = ((const uint4*)kp)[0], k1v = ((const uint4*)kp)[1];
      *(uint4*)&kT[r * 72 + seg]     = k0v;
      *(uint4*)&kT[r * 72 + seg + 8] = k1v;
#endif
      const uint4* vp = (const uint4*)(v + ((size_t)b * SS + j + r) * EMBED + hoff + seg);
      uint4 v0v = vp[0], v1v = vp[1];
      const unsigned short* pv0 = (const unsigned short*)&v0v;
      const unsigned short* pv1 = (const unsigned short*)&v1v;
#pragma unroll
      for (int i = 0; i < 8; ++i) vT[(seg + i) * 72 + r]     = pv0[i];
#pragma unroll
      for (int i = 0; i < 8; ++i) vT[(seg + 8 + i) * 72 + r] = pv1[i];
    }
    wait_async();
    __syncthreads();

    // scores: four 16-wide kv tiles, each = 2 wmma over the 64-deep head dim.
    v8f s[4];
#pragma unroll
    for (int nt = 0; nt < 4; ++nt) {
      int kr = nt * 16 + lrow;
      v16bf b0 = mk16(ld8(&kT[kr * 72 + bc]),      ld8(&kT[kr * 72 + bc + 8]));
      v16bf b1 = mk16(ld8(&kT[kr * 72 + 32 + bc]), ld8(&kT[kr * 72 + 32 + bc + 8]));
      v8f z = zero8();
      z = __builtin_amdgcn_wmma_f32_16x16x32_bf16(false, aq[0], false, b0, (short)0, z, false, false);
      z = __builtin_amdgcn_wmma_f32_16x16x32_bf16(false, aq[1], false, b1, (short)0, z, false, false);
      s[nt] = z;
    }

    // online softmax; each lane carries stats for its 8 rows (C layout).
    // 4 in-lane values combine before the 4-step width-16 lane reduction.
#pragma unroll
    for (int r = 0; r < 8; ++r) {
      float s0 = s[0][r] * 0.125f, s1 = s[1][r] * 0.125f;
      float s2 = s[2][r] * 0.125f, s3 = s[3][r] * 0.125f;
      float tm = fmaxf(fmaxf(s0, s1), fmaxf(s2, s3));
      tm = fmaxf(tm, xor16<1>(tm));
      tm = fmaxf(tm, xor16<2>(tm));
      tm = fmaxf(tm, xor16<4>(tm));
      tm = fmaxf(tm, xor16<8>(tm));
      float nm = fmaxf(mr[r], tm);
      float corr = __expf(mr[r] - nm);
      float p0 = __expf(s0 - nm), p1 = __expf(s1 - nm);
      float p2 = __expf(s2 - nm), p3 = __expf(s3 - nm);
      float rs = (p0 + p1) + (p2 + p3);
      rs += xor16<1>(rs);
      rs += xor16<2>(rs);
      rs += xor16<4>(rs);
      rs += xor16<8>(rs);
      lr[r] = lr[r] * corr + rs;
      mr[r] = nm;
#pragma unroll
      for (int dt = 0; dt < 4; ++dt) o[dt][r] *= corr;
      int prow = r + (hi ? 8 : 0);
      pS[(w * 16 + prow) * 72 + lrow]      = f2bf(p0);
      pS[(w * 16 + prow) * 72 + 16 + lrow] = f2bf(p1);
      pS[(w * 16 + prow) * 72 + 32 + lrow] = f2bf(p2);
      pS[(w * 16 + prow) * 72 + 48 + lrow] = f2bf(p3);
    }
    __syncthreads();   // make per-wave P stores visible for fragment gather

    // P (16x64) as two A fragments, V^T columns as B fragments: 8 wmma.
    v16bf ap0 = mk16(ld8(&pS[(w * 16 + lrow) * 72 + c0]),
                     ld8(&pS[(w * 16 + lrow) * 72 + c0 + 8]));
    v16bf ap1 = mk16(ld8(&pS[(w * 16 + lrow) * 72 + 32 + c0]),
                     ld8(&pS[(w * 16 + lrow) * 72 + 32 + c0 + 8]));
#pragma unroll
    for (int dt = 0; dt < 4; ++dt) {
      int d = dt * 16 + lrow;
      v16bf bv0 = mk16(ld8(&vT[d * 72 + bc]),      ld8(&vT[d * 72 + bc + 8]));
      v16bf bv1 = mk16(ld8(&vT[d * 72 + 32 + bc]), ld8(&vT[d * 72 + 32 + bc + 8]));
      o[dt] = __builtin_amdgcn_wmma_f32_16x16x32_bf16(false, ap0, false, bv0, (short)0, o[dt], false, false);
      o[dt] = __builtin_amdgcn_wmma_f32_16x16x32_bf16(false, ap1, false, bv1, (short)0, o[dt], false, false);
    }
  }

#pragma unroll
  for (int dt = 0; dt < 4; ++dt)
#pragma unroll
    for (int r = 0; r < 8; ++r) {
      int row = q0 + r + (hi ? 8 : 0);
      size_t off = ((size_t)b * SS + row) * EMBED + hoff + dt * 16 + lrow;
      ctx[off] = f2bf(o[dt][r] / lr[r]);
    }
}

// ----------------------- residual add + LayerNorm --------------------------
// one block per row; optionally also emits the bf16 copy for the next GEMM.
template<bool WB>
__global__ __launch_bounds__(256) void add_ln(
    const float* __restrict__ x, const float* __restrict__ res,
    const float* __restrict__ g, const float* __restrict__ be,
    float* __restrict__ outF, unsigned short* __restrict__ outB) {
  const int row = blockIdx.x, t = threadIdx.x;
  const float* xr = x + (size_t)row * EMBED;
  const float* rr = res + (size_t)row * EMBED;
  float vals[4], s = 0.f, s2 = 0.f;
#pragma unroll
  for (int i = 0; i < 4; ++i) {
    int c = t + i * 256;
    float vv = xr[c] + rr[c];
    vals[i] = vv; s += vv; s2 += vv * vv;
  }
#pragma unroll
  for (int off = 16; off > 0; off >>= 1) {
    s  += __shfl_xor(s,  off, 32);
    s2 += __shfl_xor(s2, off, 32);
  }
  __shared__ float redS[8], redS2[8];
  if ((t & 31) == 0) { redS[t >> 5] = s; redS2[t >> 5] = s2; }
  __syncthreads();
  s = 0.f; s2 = 0.f;
#pragma unroll
  for (int i = 0; i < 8; ++i) { s += redS[i]; s2 += redS2[i]; }
  float mean = s * (1.f / EMBED);
  float var  = s2 * (1.f / EMBED) - mean * mean;
  float inv  = rsqrtf(var + 1e-5f);
#pragma unroll
  for (int i = 0; i < 4; ++i) {
    int c = t + i * 256;
    float y = (vals[i] - mean) * inv * g[c] + be[c];
    outF[(size_t)row * EMBED + c] = y;
    if (WB) outB[(size_t)row * EMBED + c] = f2bf(y);
  }
}

// ---------------------------------------------------------------------------
extern "C" void kernel_launch(void* const* d_in, const int* in_sizes, int n_in,
                              void* d_out, int out_size, void* d_ws, size_t ws_size,
                              hipStream_t stream) {
  const float* features = (const float*)d_in[0];
  const float* Wq  = (const float*)d_in[1];
  const float* bq  = (const float*)d_in[2];
  const float* Wk  = (const float*)d_in[3];
  const float* bk  = (const float*)d_in[4];
  const float* Wv  = (const float*)d_in[5];
  const float* bv  = (const float*)d_in[6];
  const float* Wo  = (const float*)d_in[7];
  const float* bo  = (const float*)d_in[8];
  const float* g1  = (const float*)d_in[9];
  const float* be1 = (const float*)d_in[10];
  const float* W1  = (const float*)d_in[11];
  const float* b1  = (const float*)d_in[12];
  const float* W2  = (const float*)d_in[13];
  const float* b2  = (const float*)d_in[14];
  const float* g2  = (const float*)d_in[15];
  const float* be2 = (const float*)d_in[16];

  char* ws = (char*)d_ws;
  size_t cur = 0;
  auto take = [&](size_t bytes) -> char* {
    char* p = ws + cur;
    cur += (bytes + 255) & ~(size_t)255;
    return p;
  };

  unsigned short* xb  = (unsigned short*)take((size_t)MROWS * EMBED * 2);
  unsigned short* WqT = (unsigned short*)take((size_t)EMBED * EMBED * 2);
  unsigned short* WkT = (unsigned short*)take((size_t)EMBED * EMBED * 2);
  unsigned short* WvT = (unsigned short*)take((size_t)EMBED * EMBED * 2);
  unsigned short* WoT = (unsigned short*)take((size_t)EMBED * EMBED * 2);
  unsigned short* W1T = (unsigned short*)take((size_t)EMBED * FFND * 2);
  unsigned short* W2T = (unsigned short*)take((size_t)FFND * EMBED * 2);
  unsigned short* qb  = (unsigned short*)take((size_t)MROWS * EMBED * 2);
  unsigned short* kb  = (unsigned short*)take((size_t)MROWS * EMBED * 2);
  unsigned short* vb  = (unsigned short*)take((size_t)MROWS * EMBED * 2);
  unsigned short* cb  = (unsigned short*)take((size_t)MROWS * EMBED * 2);
  unsigned short* hb  = qb;  // FFN hidden (8192x4096 bf16) reuses dead q/k/v/ctx
  float* oproj = (float*)take((size_t)MROWS * EMBED * 4);
  float* ln1F  = (float*)take((size_t)MROWS * EMBED * 4);
  unsigned short* ln1B = (unsigned short*)take((size_t)MROWS * EMBED * 2);
  float* ffn2 = (float*)d_out;  // FFN2 result staged in d_out, LN2 in place

  dim3 blk(256);

  { int n = MROWS * EMBED;
    cvt_bf16<<<dim3((n + 255) / 256), blk, 0, stream>>>(features, xb, n); }
  transpose_to_bf16<<<dim3(EMBED / 32, EMBED / 32), blk, 0, stream>>>(Wq, WqT, EMBED, EMBED);
  transpose_to_bf16<<<dim3(EMBED / 32, EMBED / 32), blk, 0, stream>>>(Wk, WkT, EMBED, EMBED);
  transpose_to_bf16<<<dim3(EMBED / 32, EMBED / 32), blk, 0, stream>>>(Wv, WvT, EMBED, EMBED);
  transpose_to_bf16<<<dim3(EMBED / 32, EMBED / 32), blk, 0, stream>>>(Wo, WoT, EMBED, EMBED);
  transpose_to_bf16<<<dim3(FFND / 32, EMBED / 32), blk, 0, stream>>>(W1, W1T, EMBED, FFND);
  transpose_to_bf16<<<dim3(EMBED / 32, FFND / 32), blk, 0, stream>>>(W2, W2T, FFND, EMBED);

  dim3 gP(EMBED / 256, MROWS / 128);
  gemm_wmma<0, 1><<<gP, blk, 0, stream>>>(xb, WqT, bq, nullptr, qb, MROWS, EMBED, EMBED);
  gemm_wmma<0, 1><<<gP, blk, 0, stream>>>(xb, WkT, bk, nullptr, kb, MROWS, EMBED, EMBED);
  gemm_wmma<0, 1><<<gP, blk, 0, stream>>>(xb, WvT, bv, nullptr, vb, MROWS, EMBED, EMBED);

  flash_attn<<<dim3(SS / 128, HEADS, BB), blk, 0, stream>>>(qb, kb, vb, cb);

  gemm_wmma<0, 0><<<gP, blk, 0, stream>>>(cb, WoT, bo, oproj, nullptr, MROWS, EMBED, EMBED);
  add_ln<true><<<dim3(MROWS), blk, 0, stream>>>(oproj, features, g1, be1, ln1F, ln1B);

  gemm_wmma<1, 1><<<dim3(FFND / 256, MROWS / 128), blk, 0, stream>>>(
      ln1B, W1T, b1, nullptr, hb, MROWS, FFND, EMBED);
  gemm_wmma<0, 0><<<gP, blk, 0, stream>>>(hb, W2T, b2, ffn2, nullptr, MROWS, EMBED, FFND);
  add_ln<false><<<dim3(MROWS), blk, 0, stream>>>(ffn2, ln1F, g2, be2, (float*)d_out, nullptr);
}
